// MultiheadAttention_28638841929741
// MI455X (gfx1250) — matrix operand order repaired and checked by reading, not compiled, not run on hardware
//
#include <hip/hip_runtime.h>
#include <hip/hip_bf16.h>

typedef __attribute__((ext_vector_type(16))) __bf16 v16bf;
typedef __attribute__((ext_vector_type(8)))  __bf16 v8bf;
typedef __attribute__((ext_vector_type(8)))  float  v8f;

#define B_DIM 2
#define T_DIM 2048
#define S_DIM 2048
#define E_DIM 1024
#define H_DIM 16
#define HD_DIM 64

__device__ __forceinline__ __bf16 f2bf(float f) {
    unsigned u = __builtin_bit_cast(unsigned, f);
    unsigned r = u + 0x7FFFu + ((u >> 16) & 1u);   // round-to-nearest-even
    unsigned short h = (unsigned short)(r >> 16);
    return __builtin_bit_cast(__bf16, h);
}

// Glue two v8bf register quads into one v16bf without per-element moves.
struct bfpair { v8bf lo, hi; };
__device__ __forceinline__ v16bf combine(v8bf lo, v8bf hi) {
    bfpair p{lo, hi};
    return __builtin_bit_cast(v16bf, p);
}

// A-fragment (16x32 bf16, this lane's row): 16 contiguous K = 2*half8..+15
__device__ __forceinline__ v16bf load_a_frag(const __bf16* row, int half8) {
    return combine(*reinterpret_cast<const v8bf*>(row + 2 * half8),
                   *reinterpret_cast<const v8bf*>(row + 2 * half8 + 8));
}

// B-fragment (32x16 bf16, this lane's column): 16 contiguous K values
__device__ __forceinline__ v16bf load_b_frag(const __bf16* col) {
    return combine(*reinterpret_cast<const v8bf*>(col),
                   *reinterpret_cast<const v8bf*>(col + 8));
}

// RA = reuse-A operand-cache hint: legal only when the previous WMMA is the
// identical instruction with the same A operand (ISA §7.12 OPSEL rules).
template <bool RA>
__device__ __forceinline__ v8f wmma_bf16(v16bf a, v16bf b, v8f c) {
    return __builtin_amdgcn_wmma_f32_16x16x32_bf16(false, a, false, b, (short)0, c,
                                                   RA, false);
}

// ---------------------------------------------------------------------------
// Kernel 1: fused QKV projection.  Y = X @ W^T + b, stored bf16 head-split:
//   Y[((b*H + h)*T + t)*64 + hd]
// grid = (M/128, N/64, 3), block = 256 (8 waves). Wave w: rows m0+w*16.
// ---------------------------------------------------------------------------
__global__ __launch_bounds__(256)
void proj_qkv_kernel(const float* __restrict__ q, const float* __restrict__ k,
                     const float* __restrict__ v,
                     const float* __restrict__ Wq, const float* __restrict__ Wk,
                     const float* __restrict__ Wv,
                     const float* __restrict__ bq, const float* __restrict__ bk,
                     const float* __restrict__ bv,
                     __bf16* __restrict__ qh, __bf16* __restrict__ kh,
                     __bf16* __restrict__ vh) {
    const int which = blockIdx.z;
    const float* X    = which == 0 ? q  : which == 1 ? k  : v;
    const float* W    = which == 0 ? Wq : which == 1 ? Wk : Wv;
    const float* bias = which == 0 ? bq : which == 1 ? bk : bv;
    __bf16*      Y    = which == 0 ? qh : which == 1 ? kh : vh;

    __shared__ __bf16 Xt[128][40];
    __shared__ __bf16 Wt[64][40];

    const int tid = threadIdx.x;
    const int lane = tid & 31, wave = tid >> 5;
    const int half = lane >> 4, ln = lane & 15;
    const int m0 = blockIdx.x * 128;
    const int n0 = blockIdx.y * 64;

    v8f acc[4] = {};
    for (int kc = 0; kc < E_DIM; kc += 32) {
        {   // stage X tile 128x32 (f32 -> bf16)
            int row = tid >> 1, seg = (tid & 1) * 16;
            const float* src = X + (size_t)(m0 + row) * E_DIM + kc + seg;
            __bf16* dst = &Xt[row][seg];
#pragma unroll
            for (int j = 0; j < 16; j += 4) {
                float4 f = *reinterpret_cast<const float4*>(src + j);
                dst[j + 0] = f2bf(f.x); dst[j + 1] = f2bf(f.y);
                dst[j + 2] = f2bf(f.z); dst[j + 3] = f2bf(f.w);
            }
        }
        {   // stage W tile 64x32 (f32 -> bf16), row-major == B^T
            int row = tid >> 2, kseg = (tid & 3) * 8;
            const float* src = W + (size_t)(n0 + row) * E_DIM + kc + kseg;
            __bf16* dst = &Wt[row][kseg];
#pragma unroll
            for (int j = 0; j < 8; j += 4) {
                float4 f = *reinterpret_cast<const float4*>(src + j);
                dst[j + 0] = f2bf(f.x); dst[j + 1] = f2bf(f.y);
                dst[j + 2] = f2bf(f.z); dst[j + 3] = f2bf(f.w);
            }
        }
        __syncthreads();
        v16bf a0 = load_a_frag(&Xt[wave * 16 + ln][0], half * 8);
        v16bf b0 = load_b_frag(&Wt[0 * 16 + ln][half * 16]);
        v16bf b1 = load_b_frag(&Wt[1 * 16 + ln][half * 16]);
        v16bf b2 = load_b_frag(&Wt[2 * 16 + ln][half * 16]);
        v16bf b3 = load_b_frag(&Wt[3 * 16 + ln][half * 16]);
        acc[0] = wmma_bf16<false>(a0, b0, acc[0]);
        acc[1] = wmma_bf16<true >(a0, b1, acc[1]);
        acc[2] = wmma_bf16<true >(a0, b2, acc[2]);
        acc[3] = wmma_bf16<true >(a0, b3, acc[3]);
        __syncthreads();
    }
    // bias + store head-split bf16
#pragma unroll
    for (int c = 0; c < 4; ++c) {
        int n = n0 + c * 16 + ln;
        float bb = bias[n];
        int h = n >> 6, hd = n & 63;
#pragma unroll
        for (int r = 0; r < 8; ++r) {
            int m = m0 + wave * 16 + half * 8 + r;
            int b = m >> 11;            // / T_DIM
            int t = m & (T_DIM - 1);
            Y[(((size_t)(b * H_DIM + h) * T_DIM) + t) * HD_DIM + hd] =
                f2bf(acc[c][r] + bb);
        }
    }
}

// ---------------------------------------------------------------------------
// Kernel 2: flash attention. grid = (T/128, B*H), block = 256 (8 waves).
// Wave w handles 16 query rows; K/V chunks (32 cols of S) shared in LDS.
// ---------------------------------------------------------------------------
__global__ __launch_bounds__(256)
void flash_attn_kernel(const __bf16* __restrict__ qh, const __bf16* __restrict__ kh,
                       const __bf16* __restrict__ vh, const int* __restrict__ mask,
                       __bf16* __restrict__ attn_out) {
    const int bh = blockIdx.y;            // b*H + h
    const int b = bh >> 4, h = bh & 15;
    const int tid = threadIdx.x;
    const int lane = tid & 31, wave = tid >> 5;
    const int half = lane >> 4, ln = lane & 15;
    const int qbase = blockIdx.x * 128 + wave * 16;

    __shared__ __bf16 Kt[32][72];         // [s_local][d]
    __shared__ __bf16 Vt[64][40];         // [d][s_local]  (transposed)
    __shared__ __bf16 Pb[8][16][40];      // per-wave P round-trip buffer

    const __bf16* Qrow = qh + (((size_t)bh * T_DIM) + qbase + ln) * HD_DIM;
    v16bf aq0 = load_a_frag(Qrow, half * 8);        // d = 0..31
    v16bf aq1 = load_a_frag(Qrow + 32, half * 8);   // d = 32..63

    v8f o[4] = {};
    float mrun[8], lrun[8];
#pragma unroll
    for (int r = 0; r < 8; ++r) { mrun[r] = -1e30f; lrun[r] = 0.f; }
    const float scale = 0.125f;           // 1/sqrt(64)

    const int ldrow = tid >> 3, lddg = (tid & 7) * 8;
    const size_t kvbase = (size_t)bh * S_DIM;

    for (int s0 = 0; s0 < S_DIM; s0 += 32) {
        __syncthreads();
        {   // K chunk 32x64 bf16 -> Kt
            const __bf16* src = kh + (kvbase + s0 + ldrow) * HD_DIM + lddg;
            *reinterpret_cast<uint4*>(&Kt[ldrow][lddg]) =
                *reinterpret_cast<const uint4*>(src);
        }
        {   // V chunk 32x64 bf16 -> Vt transposed
            const __bf16* src = vh + (kvbase + s0 + ldrow) * HD_DIM + lddg;
            v8bf vv = *reinterpret_cast<const v8bf*>(src);
#pragma unroll
            for (int j = 0; j < 8; ++j) Vt[lddg + j][ldrow] = vv[j];
        }
        // prefetch next chunk's K/V lines into cache (global_prefetch_b8)
        if (s0 + 32 < S_DIM) {
            __builtin_prefetch(kh + (kvbase + s0 + 32 + ldrow) * HD_DIM + lddg, 0, 1);
            __builtin_prefetch(vh + (kvbase + s0 + 32 + ldrow) * HD_DIM + lddg, 0, 1);
        }
        __syncthreads();

        // S = Q K^T for two 16-wide column tiles
        v8f st[2];
#pragma unroll
        for (int sc = 0; sc < 2; ++sc) {
            v8f acc = {};
            v16bf bk0 = load_b_frag(&Kt[sc * 16 + ln][half * 16]);
            v16bf bk1 = load_b_frag(&Kt[sc * 16 + ln][32 + half * 16]);
            acc = wmma_bf16<false>(aq0, bk0, acc);
            acc = wmma_bf16<false>(aq1, bk1, acc);
            int scol = s0 + sc * 16 + ln;
            const int* mp = mask + (size_t)(qbase + half * 8) * S_DIM + scol;
#pragma unroll
            for (int r = 0; r < 8; ++r) {
                float x = acc[r] * scale;
                if (mp[(size_t)r * S_DIM] != 0) x = -10000.f;
                acc[r] = x;
            }
            st[sc] = acc;
        }

        // online softmax: row stats live per-lane for rows half*8+r
        float rmax[8];
#pragma unroll
        for (int r = 0; r < 8; ++r) rmax[r] = fmaxf(st[0][r], st[1][r]);
#pragma unroll
        for (int off = 1; off < 16; off <<= 1)
#pragma unroll
            for (int r = 0; r < 8; ++r)
                rmax[r] = fmaxf(rmax[r], __shfl_xor(rmax[r], off, 16));

        float alpha[8], rsum[8];
#pragma unroll
        for (int r = 0; r < 8; ++r) {
            float mn = fmaxf(mrun[r], rmax[r]);
            alpha[r] = __expf(mrun[r] - mn);
            mrun[r] = mn;
            float p0 = __expf(st[0][r] - mn);
            float p1 = __expf(st[1][r] - mn);
            st[0][r] = p0; st[1][r] = p1;
            rsum[r] = p0 + p1;
        }
#pragma unroll
        for (int off = 1; off < 16; off <<= 1)
#pragma unroll
            for (int r = 0; r < 8; ++r) rsum[r] += __shfl_xor(rsum[r], off, 16);
#pragma unroll
        for (int r = 0; r < 8; ++r) lrun[r] = lrun[r] * alpha[r] + rsum[r];
#pragma unroll
        for (int c = 0; c < 4; ++c)
#pragma unroll
            for (int r = 0; r < 8; ++r) o[c][r] *= alpha[r];

        // C-layout P -> LDS -> A-layout (same-wave round trip)
#pragma unroll
        for (int sc = 0; sc < 2; ++sc)
#pragma unroll
            for (int r = 0; r < 8; ++r)
                Pb[wave][half * 8 + r][sc * 16 + ln] = f2bf(st[sc][r]);
        asm volatile("s_wait_dscnt 0" ::: "memory");

        v16bf ap = load_a_frag(&Pb[wave][ln][0], half * 8);
        v16bf bv0 = load_b_frag(&Vt[0 * 16 + ln][half * 16]);
        v16bf bv1 = load_b_frag(&Vt[1 * 16 + ln][half * 16]);
        v16bf bv2 = load_b_frag(&Vt[2 * 16 + ln][half * 16]);
        v16bf bv3 = load_b_frag(&Vt[3 * 16 + ln][half * 16]);
        o[0] = wmma_bf16<false>(ap, bv0, o[0]);
        o[1] = wmma_bf16<true >(ap, bv1, o[1]);
        o[2] = wmma_bf16<true >(ap, bv2, o[2]);
        o[3] = wmma_bf16<true >(ap, bv3, o[3]);
    }

    // normalize + store bf16 at [b][t][h*64 + d]
#pragma unroll
    for (int c = 0; c < 4; ++c)
#pragma unroll
        for (int r = 0; r < 8; ++r) {
            int t = qbase + half * 8 + r;
            attn_out[(((size_t)b * T_DIM + t) * E_DIM) + h * 64 + c * 16 + ln] =
                f2bf(o[c][r] / lrun[r]);
        }
}

// ---------------------------------------------------------------------------
// Kernel 3: output projection. X (bf16, [B*T, E]) @ Wo^T + bo -> f32 out.
// grid = (M/128, N/64), block 256.
// ---------------------------------------------------------------------------
__global__ __launch_bounds__(256)
void proj_out_kernel(const __bf16* __restrict__ X, const float* __restrict__ W,
                     const float* __restrict__ bias, float* __restrict__ out) {
    __shared__ __bf16 Xt[128][40];
    __shared__ __bf16 Wt[64][40];

    const int tid = threadIdx.x;
    const int lane = tid & 31, wave = tid >> 5;
    const int half = lane >> 4, ln = lane & 15;
    const int m0 = blockIdx.x * 128;
    const int n0 = blockIdx.y * 64;

    v8f acc[4] = {};
    for (int kc = 0; kc < E_DIM; kc += 32) {
        {   // stage X tile 128x32 (already bf16): two 16B vector copies
            int row = tid >> 1, seg = (tid & 1) * 16;
            const __bf16* src = X + (size_t)(m0 + row) * E_DIM + kc + seg;
            *reinterpret_cast<uint4*>(&Xt[row][seg]) =
                *reinterpret_cast<const uint4*>(src);
            *reinterpret_cast<uint4*>(&Xt[row][seg + 8]) =
                *reinterpret_cast<const uint4*>(src + 8);
        }
        {   // stage Wo tile 64x32 (f32 -> bf16)
            int row = tid >> 2, kseg = (tid & 3) * 8;
            const float* src = W + (size_t)(n0 + row) * E_DIM + kc + kseg;
            __bf16* dst = &Wt[row][kseg];
#pragma unroll
            for (int j = 0; j < 8; j += 4) {
                float4 f = *reinterpret_cast<const float4*>(src + j);
                dst[j + 0] = f2bf(f.x); dst[j + 1] = f2bf(f.y);
                dst[j + 2] = f2bf(f.z); dst[j + 3] = f2bf(f.w);
            }
        }
        __syncthreads();
        v16bf a0 = load_a_frag(&Xt[wave * 16 + ln][0], half * 8);
        v16bf b0 = load_b_frag(&Wt[0 * 16 + ln][half * 16]);
        v16bf b1 = load_b_frag(&Wt[1 * 16 + ln][half * 16]);
        v16bf b2 = load_b_frag(&Wt[2 * 16 + ln][half * 16]);
        v16bf b3 = load_b_frag(&Wt[3 * 16 + ln][half * 16]);
        acc[0] = wmma_bf16<false>(a0, b0, acc[0]);
        acc[1] = wmma_bf16<true >(a0, b1, acc[1]);
        acc[2] = wmma_bf16<true >(a0, b2, acc[2]);
        acc[3] = wmma_bf16<true >(a0, b3, acc[3]);
        __syncthreads();
    }
#pragma unroll
    for (int c = 0; c < 4; ++c) {
        int n = n0 + c * 16 + ln;
        float bb = bias[n];
#pragma unroll
        for (int r = 0; r < 8; ++r) {
            int m = m0 + wave * 16 + half * 8 + r;
            out[(size_t)m * E_DIM + n] = acc[c][r] + bb;
        }
    }
}

// ---------------------------------------------------------------------------
extern "C" void kernel_launch(void* const* d_in, const int* in_sizes, int n_in,
                              void* d_out, int out_size, void* d_ws, size_t ws_size,
                              hipStream_t stream) {
    const float* q    = (const float*)d_in[0];
    const float* k    = (const float*)d_in[1];
    const float* v    = (const float*)d_in[2];
    const int*   mask = (const int*)  d_in[3];
    const float* Wq   = (const float*)d_in[4];
    const float* bq   = (const float*)d_in[5];
    const float* Wk   = (const float*)d_in[6];
    const float* bk   = (const float*)d_in[7];
    const float* Wv   = (const float*)d_in[8];
    const float* bv   = (const float*)d_in[9];
    const float* Wo   = (const float*)d_in[10];
    const float* bo   = (const float*)d_in[11];
    float* out = (float*)d_out;

    const size_t NEL = (size_t)B_DIM * H_DIM * T_DIM * HD_DIM;  // 4,194,304
    __bf16* qh   = (__bf16*)d_ws;
    __bf16* kh   = qh + NEL;
    __bf16* vh   = kh + NEL;
    __bf16* attn = vh + NEL;   // [B*T, E] bf16

    // 1) fused QKV projection (z selects q/k/v)
    proj_qkv_kernel<<<dim3((B_DIM * T_DIM) / 128, E_DIM / 64, 3), 256, 0, stream>>>(
        q, k, v, Wq, Wk, Wv, bq, bk, bv, qh, kh, vh);

    // 2) flash attention over S in 32-wide chunks
    flash_attn_kernel<<<dim3(T_DIM / 128, B_DIM * H_DIM), 256, 0, stream>>>(
        qh, kh, vh, mask, attn);

    // 3) output projection -> f32
    proj_out_kernel<<<dim3((B_DIM * T_DIM) / 128, E_DIM / 64), 256, 0, stream>>>(
        attn, Wo, bo, out);
}
